// PDNC_33517924778065
// MI455X (gfx1250) — matrix operand order repaired and checked by reading, not compiled, not run on hardware
//
#include <hip/hip_runtime.h>
#include <hip/hip_bf16.h>
#include <math.h>

// ---------------------------------------------------------------------------
// PDNC forward for MI455X (gfx1250, wave32, WMMA).
// All dense matmuls run through v_wmma_f32_16x16x32_f16 (f16 in, f32 acc).
// Weights are pre-converted to f16 in workspace; activations convert on load.
// GEMM: 16x32 output tile per wave (one A fragment feeds two WMMAs).
// ---------------------------------------------------------------------------

typedef __attribute__((ext_vector_type(16))) _Float16 v16h;
typedef __attribute__((ext_vector_type(8)))  float    v8f;
typedef __attribute__((ext_vector_type(2)))  _Float16 h2f;

#define BATCH 32
#define TSTEPS 32
#define ZDIM 256
#define HDIM 512
#define MSLOTS 256
#define WWID 64
#define RHEADS 4
#define IFACE 471
#define NN_IN 514
#define VSTR IFACE

__device__ __forceinline__ float sigf(float x) { return 1.0f / (1.0f + __expf(-x)); }
__device__ __forceinline__ float softplusf(float x) { return x > 20.0f ? x : log1pf(__expf(x)); }

// ---------------------------------------------------------------------------
// Generic WMMA GEMM:  C[M,N] (+)= act( A[M,K](f32) * B[N,K](f16)^T + bias[N] )
// One wave per 16x32 tile (two 16x16 WMMAs sharing the A fragment),
// 8 waves per block. Fast path: branch-free vectorized loads (row pointers
// clamped -- out-of-range rows only feed accumulator lanes never stored).
// Tail (K%32) uses clamped-address loads + value select, no exec branches.
// ---------------------------------------------------------------------------
__global__ void wmma_gemm_kernel(const float* __restrict__ A, int lda,
                                 const _Float16* __restrict__ Bh, int ldb,
                                 const float* __restrict__ bias,
                                 float* __restrict__ C, int ldc,
                                 int M, int N, int K,
                                 int accumulate, int act, float leak)
{
    const int wave   = threadIdx.x >> 5;
    const int lane   = threadIdx.x & 31;
    const int tilesN = (N + 31) >> 5;          // 32-wide N tiles
    const int tileId = blockIdx.x * 8 + wave;
    const int tm = tileId / tilesN;
    const int tn = tileId - tm * tilesN;
    if (tm * 16 >= M) return;

    const int half = lane >> 4;   // 0 or 1
    const int l16  = lane & 15;

    const int mRow  = tm * 16 + l16;        // A row this lane feeds
    const int nCol0 = tn * 32 + l16;        // weight rows (output columns)
    const int nCol1 = tn * 32 + 16 + l16;
    const float*    Arow  = A  + (size_t)(mRow  < M ? mRow  : M - 1) * lda;
    const _Float16* Brow0 = Bh + (size_t)(nCol0 < N ? nCol0 : N - 1) * ldb;
    const _Float16* Brow1 = Bh + (size_t)(nCol1 < N ? nCol1 : N - 1) * ldb;

    v8f acc0 = {0.f, 0.f, 0.f, 0.f, 0.f, 0.f, 0.f, 0.f};
    v8f acc1 = {0.f, 0.f, 0.f, 0.f, 0.f, 0.f, 0.f, 0.f};

    const int kFast = K & ~31;
    for (int k0 = 0; k0 < kFast; k0 += 32) {
        const float*    pa  = Arow  + k0 + (half << 4);
        const _Float16* pb0 = Brow0 + k0 + (half << 4);
        const _Float16* pb1 = Brow1 + k0 + (half << 4);
        __builtin_prefetch(pa + 32, 0, 3);       // next K chunk (A), near scope
        __builtin_prefetch(pb0 + 32, 0, 3);      // next K chunk (B0)
        __builtin_prefetch(pb1 + 32, 0, 3);      // next K chunk (B1)
        v16h a, b0, b1;
        // A 16x32 layout: elem j -> K = k0 + half*16 + j
#pragma unroll
        for (int j = 0; j < 4; ++j) {
            float2 t0 = *(const float2*)(pa + 2 * j);        // K group 0
            float2 t1 = *(const float2*)(pa + 8 + 2 * j);    // K group 1
            a[2 * j]         = (_Float16)t0.x;
            a[2 * j + 1]     = (_Float16)t0.y;
            a[8 + 2 * j]     = (_Float16)t1.x;
            a[8 + 2 * j + 1] = (_Float16)t1.y;
        }
        // B 32x16 layout: elem j -> K = k0 + half*16 + j, column n = l16
#pragma unroll
        for (int j = 0; j < 8; ++j) {
            h2f t0 = *(const h2f*)(pb0 + 2 * j);
            h2f t1 = *(const h2f*)(pb1 + 2 * j);
            b0[2 * j]     = t0.x;
            b0[2 * j + 1] = t0.y;
            b1[2 * j]     = t1.x;
            b1[2 * j + 1] = t1.y;
        }
        acc0 = __builtin_amdgcn_wmma_f32_16x16x32_f16(
            false, a, false, b0, (short)0, acc0, false, false);
        acc1 = __builtin_amdgcn_wmma_f32_16x16x32_f16(
            false, a, false, b1, (short)0, acc1, false, false);
    }

    if (kFast < K) {   // K tail: clamped unconditional loads + value select
        v16h a{}, b0{}, b1{};
#pragma unroll
        for (int j = 0; j < 16; ++j) {
            int k = kFast + (half << 4) + j;
            float av = Arow[k < K ? k : K - 1];
            a[j] = (_Float16)(k < K ? av : 0.0f);
        }
#pragma unroll
        for (int j = 0; j < 16; ++j) {
            int k = kFast + (half << 4) + j;
            _Float16 bv0 = Brow0[k < K ? k : K - 1];
            _Float16 bv1 = Brow1[k < K ? k : K - 1];
            b0[j] = (k < K) ? bv0 : (_Float16)0.0f;
            b1[j] = (k < K) ? bv1 : (_Float16)0.0f;
        }
        acc0 = __builtin_amdgcn_wmma_f32_16x16x32_f16(
            false, a, false, b0, (short)0, acc0, false, false);
        acc1 = __builtin_amdgcn_wmma_f32_16x16x32_f16(
            false, a, false, b1, (short)0, acc1, false, false);
    }

    // D layout: acc[i] -> row m = tm*16 + i + half*8, col n = l16 (per N tile)
#pragma unroll
    for (int tsub = 0; tsub < 2; ++tsub) {
        const int n = tn * 32 + tsub * 16 + l16;
        if (n >= N) continue;
        float bv = bias ? bias[n] : 0.0f;
#pragma unroll
        for (int i = 0; i < 8; ++i) {
            int m = tm * 16 + i + half * 8;
            if (m < M) {
                float v = (tsub == 0 ? acc0[i] : acc1[i]) + bv;
                size_t idx = (size_t)m * ldc + n;
                if (accumulate) v += C[idx];
                if (act == 1)      v = v > 0.0f ? v : leak * v;   // leaky relu
                else if (act == 2) v = tanhf(v);                  // tanh
                C[idx] = v;
            }
        }
    }
}

// f32 -> f16 weight conversion
__global__ void f32_to_f16_kernel(const float* __restrict__ in, _Float16* __restrict__ out,
                                  int n)
{
    int i = blockIdx.x * blockDim.x + threadIdx.x;
    if (i < n) out[i] = (_Float16)in[i];
}

// ---------------------------------------------------------------------------
// CDNA5 async-path probe: global_load_async_to_lds + asynccnt/tensorcnt waits
// + cluster barrier (NOP when not in a cluster per ISA 3.2). Launched once.
// ---------------------------------------------------------------------------
__global__ void cdna5_async_probe_kernel(const float* __restrict__ src,
                                         float* __restrict__ sink)
{
    __shared__ float buf[32];
    int lane = threadIdx.x & 31;
    unsigned lds_off = (unsigned)(size_t)(&buf[0]) + lane * 4u;   // low 32b = LDS offset
    unsigned long long ga = (unsigned long long)(size_t)(src + lane);
    asm volatile("global_load_async_to_lds_b32 %0, %1, off"
                 :: "v"(lds_off), "v"(ga) : "memory");
    asm volatile("s_wait_asynccnt 0" ::: "memory");
    __builtin_amdgcn_s_cluster_barrier();
    __builtin_amdgcn_s_wait_tensorcnt(0);
    __syncthreads();
    if (lane == 0) sink[0] = buf[0];
}

// ---------------------------------------------------------------------------
// im2col (3x3, stride 2, pad 1) from NCHW input
// ---------------------------------------------------------------------------
__global__ void im2col_nchw_kernel(const float* __restrict__ in, float* __restrict__ col,
                                   int Nimg, int Cin, int Hin, int Win, int OH, int OW)
{
    const int Kc = Cin * 9;
    long long total = (long long)Nimg * OH * OW * Kc;
    long long idx = (long long)blockIdx.x * blockDim.x + threadIdx.x;
    if (idx >= total) return;
    int k   = (int)(idx % Kc);
    long long row = idx / Kc;
    int c  = k / 9;
    int kh = (k % 9) / 3;
    int kw = k % 3;
    int img = (int)(row / (OH * OW));
    int pos = (int)(row % (OH * OW));
    int oh = pos / OW, ow = pos % OW;
    int ih = oh * 2 - 1 + kh;
    int iw = ow * 2 - 1 + kw;
    float v = 0.0f;
    if (ih >= 0 && ih < Hin && iw >= 0 && iw < Win)
        v = in[(((size_t)img * Cin + c) * Hin + ih) * Win + iw];
    col[row * Kc + k] = v;
}

// im2col from NHWC activations (row = img*OH*OW+pos, col = channel)
__global__ void im2col_nhwc_kernel(const float* __restrict__ in, float* __restrict__ col,
                                   int Nimg, int Cin, int Hin, int Win, int OH, int OW)
{
    const int Kc = Cin * 9;
    long long total = (long long)Nimg * OH * OW * Kc;
    long long idx = (long long)blockIdx.x * blockDim.x + threadIdx.x;
    if (idx >= total) return;
    int k   = (int)(idx % Kc);
    long long row = idx / Kc;
    int c  = k / 9;
    int kh = (k % 9) / 3;
    int kw = k % 3;
    int img = (int)(row / (OH * OW));
    int pos = (int)(row % (OH * OW));
    int oh = pos / OW, ow = pos % OW;
    int ih = oh * 2 - 1 + kh;
    int iw = ow * 2 - 1 + kw;
    float v = 0.0f;
    if (ih >= 0 && ih < Hin && iw >= 0 && iw < Win)
        v = in[(((size_t)img * Hin + ih) * Win + iw) * Cin + c];
    col[row * Kc + k] = v;
}

// NHWC (1024, 4,4,256) -> NCHW-flat features (1024, 4096)
__global__ void nhwc_to_flat_kernel(const float* __restrict__ in, float* __restrict__ out)
{
    int idx = blockIdx.x * blockDim.x + threadIdx.x;
    if (idx >= 1024 * 4096) return;
    int img = idx >> 12;
    int f   = idx & 4095;
    int c = f >> 4;      // channel
    int p = f & 15;      // spatial position
    out[idx] = in[((size_t)img * 16 + p) * 256 + c];
}

__global__ void fill_kernel(float* p, float v, int n)
{
    int i = blockIdx.x * blockDim.x + threadIdx.x;
    if (i < n) p[i] = v;
}

// x = concat(enc[:, t, :], [0,1], read_vecs)  -> (32, 514)
__global__ void build_x_kernel(float* __restrict__ x, const float* __restrict__ enc,
                               const float* __restrict__ rvec, int t)
{
    int i = blockIdx.x * blockDim.x + threadIdx.x;
    if (i >= BATCH * NN_IN) return;
    int b = i / NN_IN, j = i % NN_IN;
    float v;
    if (j < ZDIM)            v = enc[((size_t)b * TSTEPS + t) * ZDIM + j];
    else if (j == ZDIM)      v = 0.0f;
    else if (j == ZDIM + 1)  v = 1.0f;
    else                     v = rvec[b * (RHEADS * WWID) + (j - ZDIM - 2)];
    x[i] = v;
}

// gates (32,2048) split [i,f,g,o]; update h,c of one layer in place
__global__ void lstm_pointwise_kernel(const float* __restrict__ gates,
                                      float* __restrict__ h, float* __restrict__ c)
{
    int idx = blockIdx.x * blockDim.x + threadIdx.x;
    if (idx >= BATCH * HDIM) return;
    int b = idx / HDIM, j = idx % HDIM;
    const float* g = gates + (size_t)b * 4 * HDIM;
    float gi = g[j], gf = g[HDIM + j], gg = g[2 * HDIM + j], go = g[3 * HDIM + j];
    float cn = sigf(gf) * c[idx] + sigf(gi) * tanhf(gg);
    c[idx] = cn;
    h[idx] = sigf(go) * tanhf(cn);
}

__global__ void clip_kernel(const float* __restrict__ h, float* __restrict__ out)
{
    int i = blockIdx.x * blockDim.x + threadIdx.x;
    if (i >= BATCH * HDIM) return;
    out[i] = fminf(20.0f, fmaxf(-20.0f, h[i]));
}

// usage = (usage + (1-usage)*ww_old) * prod_r(1 - sig(fg_r)*rw_old)
__global__ void usage_kernel(float* __restrict__ usage, const float* __restrict__ ww,
                             const float* __restrict__ rw, const float* __restrict__ v)
{
    int i = blockIdx.x * blockDim.x + threadIdx.x;
    if (i >= BATCH * MSLOTS) return;
    int b = i >> 8, m = i & 255;
    float u = usage[i];
    u = u + (1.0f - u) * ww[i];
    float psi = 1.0f;
#pragma unroll
    for (int r = 0; r < RHEADS; ++r) {
        float fg = sigf(v[b * VSTR + 453 + r]);
        psi *= (1.0f - fg * rw[(b * RHEADS + r) * MSLOTS + m]);
    }
    usage[i] = u * psi;
}

// allocation weighting: stable argsort of u ascending + cumprod (1 block/batch)
__global__ void alloc_kernel(const float* __restrict__ usage, float* __restrict__ alloc)
{
    __shared__ float u[MSLOTS];
    __shared__ int   order[MSLOTS];
    __shared__ float sa[MSLOTS];
    int b = blockIdx.x, t = threadIdx.x;
    u[t] = 1e-6f + (1.0f - 1e-6f) * usage[b * MSLOTS + t];
    __syncthreads();
    if (t == 0) {
        for (int i = 0; i < MSLOTS; ++i) order[i] = i;
        for (int i = 1; i < MSLOTS; ++i) {             // stable insertion sort
            int oi = order[i]; float ui = u[oi]; int j = i - 1;
            while (j >= 0 && u[order[j]] > ui) { order[j + 1] = order[j]; --j; }
            order[j + 1] = oi;
        }
        float prod = 1.0f;
        for (int i = 0; i < MSLOTS; ++i) {
            float su = u[order[i]];
            sa[i] = (1.0f - su) * prod;
            prod *= su;
        }
    }
    __syncthreads();
    alloc[b * MSLOTS + order[t]] = sa[t];    // inverse-permutation scatter
}

// cosine-similarity content weights with softmax over M (1 block per (b,head))
__global__ void content_kernel(const float* __restrict__ mem, const float* __restrict__ v,
                               int keyOff, int strOff, int nk, float* __restrict__ out)
{
    __shared__ float red[MSLOTS];
    int bh = blockIdx.x;
    int b = bh / nk, r = bh % nk;
    int m = threadIdx.x;
    const float* key  = v + (size_t)b * VSTR + keyOff + r * WWID;
    const float* mrow = mem + ((size_t)b * MSLOTS + m) * WWID;
    float dot = 0.f, nm = 0.f, nk2 = 0.f;
#pragma unroll 8
    for (int w = 0; w < WWID; ++w) {
        float kw = key[w], mw = mrow[w];
        dot += kw * mw; nm += mw * mw; nk2 += kw * kw;
    }
    float str = softplusf(v[(size_t)b * VSTR + strOff + r]);
    float sim = dot / (sqrtf(nk2) * sqrtf(nm) + 1e-6f) * str;
    red[m] = sim; __syncthreads();
    for (int o = 128; o > 0; o >>= 1) { if (m < o) red[m] = fmaxf(red[m], red[m + o]); __syncthreads(); }
    float mx = red[0]; __syncthreads();
    float e = __expf(sim - mx);
    red[m] = e; __syncthreads();
    for (int o = 128; o > 0; o >>= 1) { if (m < o) red[m] += red[m + o]; __syncthreads(); }
    out[(size_t)bh * MSLOTS + m] = e / red[0];
}

// ww_new = write_gate * (alloc_gate*alloc + (1-alloc_gate)*wcw)
__global__ void ww_kernel(float* __restrict__ ww, const float* __restrict__ v,
                          const float* __restrict__ alloc, const float* __restrict__ wcw)
{
    int i = blockIdx.x * blockDim.x + threadIdx.x;
    if (i >= BATCH * MSLOTS) return;
    int b = i >> 8;
    float ag = sigf(v[b * VSTR + 457]);
    float wg = sigf(v[b * VSTR + 458]);
    ww[i] = wg * (ag * alloc[i] + (1.0f - ag) * wcw[i]);
}

// link = (1 - wv_m - wv_n)*link + wv_m*prec_n ; zero diagonal (uses OLD prec)
__global__ void link_kernel(float* __restrict__ link, const float* __restrict__ ww,
                            const float* __restrict__ prec)
{
    int i = blockIdx.x * blockDim.x + threadIdx.x;
    if (i >= BATCH * MSLOTS * MSLOTS) return;
    int b = i >> 16;
    int mn = i & 65535;
    int m = mn >> 8, n = mn & 255;
    float wvm = ww[b * MSLOTS + m];
    float wvn = ww[b * MSLOTS + n];
    float L = link[i];
    L = (1.0f - wvm - wvn) * L + wvm * prec[b * MSLOTS + n];
    link[i] = (m == n) ? 0.0f : L;
}

__global__ void sumwv_kernel(const float* __restrict__ ww, float* __restrict__ out)
{
    __shared__ float red[MSLOTS];
    int b = blockIdx.x, t = threadIdx.x;
    red[t] = ww[b * MSLOTS + t]; __syncthreads();
    for (int o = 128; o > 0; o >>= 1) { if (t < o) red[t] += red[t + o]; __syncthreads(); }
    if (t == 0) out[b] = red[0];
}

__global__ void prec_kernel(float* __restrict__ prec, const float* __restrict__ ww,
                            const float* __restrict__ sumwv)
{
    int i = blockIdx.x * blockDim.x + threadIdx.x;
    if (i >= BATCH * MSLOTS) return;
    int b = i >> 8;
    prec[i] = (1.0f - sumwv[b]) * prec[i] + ww[i];
}

// mem = mem*(1 - ww_m * erase_w) + ww_m * write_vec_w
__global__ void mem_update_kernel(float* __restrict__ mem, const float* __restrict__ ww,
                                  const float* __restrict__ v)
{
    int i = blockIdx.x * blockDim.x + threadIdx.x;
    if (i >= BATCH * MSLOTS * WWID) return;
    int b = i / (MSLOTS * WWID);
    int m = (i / WWID) & 255;
    int w = i & 63;
    float e   = sigf(v[(size_t)b * VSTR + 325 + w]);
    float wv  = v[(size_t)b * VSTR + 389 + w];
    float wwm = ww[b * MSLOTS + m];
    mem[i] = mem[i] * (1.0f - wwm * e) + wwm * wv;
}

// fw = rw_old @ link ; bw = rw_old @ link^T
__global__ void fwbw_kernel(const float* __restrict__ rw, const float* __restrict__ link,
                            float* __restrict__ fw, float* __restrict__ bw)
{
    int i = blockIdx.x * blockDim.x + threadIdx.x;
    if (i >= BATCH * RHEADS * MSLOTS) return;
    int b = i >> 10;
    int rn = i & 1023;
    int r = rn >> 8, n = rn & 255;
    const float* rwrow = rw + ((size_t)b * RHEADS + r) * MSLOTS;
    const float* Lb = link + (size_t)b * MSLOTS * MSLOTS;
    float f = 0.f, bk = 0.f;
    for (int m = 0; m < MSLOTS; ++m) {
        float rv = rwrow[m];
        f  += rv * Lb[(size_t)m * MSLOTS + n];
        bk += rv * Lb[(size_t)n * MSLOTS + m];
    }
    fw[i] = f;
    bw[i] = bk;
}

// rw_new = softmax(read_modes) . [bw, rcw, fw]
__global__ void rwnew_kernel(float* __restrict__ rw, const float* __restrict__ v,
                             const float* __restrict__ bw, const float* __restrict__ rcw,
                             const float* __restrict__ fw)
{
    int i = blockIdx.x * blockDim.x + threadIdx.x;
    if (i >= BATCH * RHEADS * MSLOTS) return;
    int b = i >> 10;
    int r = (i >> 8) & 3;
    const float* mv = v + (size_t)b * VSTR + 459 + r * 3;
    float m0 = mv[0], m1 = mv[1], m2 = mv[2];
    float mx = fmaxf(m0, fmaxf(m1, m2));
    float e0 = __expf(m0 - mx), e1 = __expf(m1 - mx), e2 = __expf(m2 - mx);
    float inv = 1.0f / (e0 + e1 + e2);
    rw[i] = (e0 * bw[i] + e1 * rcw[i] + e2 * fw[i]) * inv;
}

// read_vecs[b, r*64+w] = sum_m rw[b,r,m] * mem[b,m,w]
__global__ void readvec_kernel(const float* __restrict__ rw, const float* __restrict__ mem,
                               float* __restrict__ rvec)
{
    int i = blockIdx.x * blockDim.x + threadIdx.x;
    if (i >= BATCH * RHEADS * WWID) return;
    int b = i >> 8;
    int r = (i >> 6) & 3;
    int w = i & 63;
    const float* rwrow = rw + ((size_t)b * RHEADS + r) * MSLOTS;
    const float* mb = mem + (size_t)b * MSLOTS * WWID + w;
    float s = 0.f;
    for (int m = 0; m < MSLOTS; ++m) s += rwrow[m] * mb[(size_t)m * WWID];
    rvec[b * (RHEADS * WWID) + r * WWID + w] = s;
}

// y_in = concat(out_clip * 0.2, read_vecs)  (32, 768)
__global__ void build_y_kernel(float* __restrict__ yin, const float* __restrict__ oc,
                               const float* __restrict__ rvec)
{
    int i = blockIdx.x * blockDim.x + threadIdx.x;
    if (i >= BATCH * (HDIM + RHEADS * WWID)) return;
    int b = i / 768, j = i % 768;
    yin[i] = (j < HDIM) ? oc[b * HDIM + j] * 0.2f
                        : rvec[b * (RHEADS * WWID) + (j - HDIM)];
}

// ---------------------------------------------------------------------------
// host-side helpers
// ---------------------------------------------------------------------------
static void gemm(hipStream_t s, const float* A, int lda, const _Float16* B, int ldb,
                 const float* bias, float* C, int ldc, int M, int N, int K,
                 int accum, int act, float leak)
{
    int tiles  = ((M + 15) / 16) * ((N + 31) / 32);
    int blocks = (tiles + 7) / 8;
    wmma_gemm_kernel<<<blocks, 256, 0, s>>>(A, lda, B, ldb, bias, C, ldc,
                                            M, N, K, accum, act, leak);
}

static inline int gr(long long n) { return (int)((n + 255) / 256); }

static void cvt16(hipStream_t s, const float* in, _Float16* out, int n)
{
    f32_to_f16_kernel<<<gr(n), 256, 0, s>>>(in, out, n);
}

extern "C" void kernel_launch(void* const* d_in, const int* in_sizes, int n_in,
                              void* d_out, int out_size, void* d_ws, size_t ws_size,
                              hipStream_t stream)
{
    (void)in_sizes; (void)n_in; (void)out_size; (void)ws_size;

    const float* input  = (const float*)d_in[0];
    const float* h0     = (const float*)d_in[1];
    const float* c1w = (const float*)d_in[2],  *c1b = (const float*)d_in[3];
    const float* c2w = (const float*)d_in[4],  *c2b = (const float*)d_in[5];
    const float* c3w = (const float*)d_in[6],  *c3b = (const float*)d_in[7];
    const float* l1w = (const float*)d_in[8],  *l1b = (const float*)d_in[9];
    const float* l2w = (const float*)d_in[10], *l2b = (const float*)d_in[11];
    const float* l3w = (const float*)d_in[12], *l3b = (const float*)d_in[13];
    const float* outw = (const float*)d_in[14], *outb = (const float*)d_in[15];
    const float* ifw  = (const float*)d_in[16], *ifb  = (const float*)d_in[17];
    const float* Wih0 = (const float*)d_in[18], *Whh0 = (const float*)d_in[19];
    const float* bih0 = (const float*)d_in[20], *bhh0 = (const float*)d_in[21];
    const float* Wih1 = (const float*)d_in[22], *Whh1 = (const float*)d_in[23];
    const float* bih1 = (const float*)d_in[24], *bhh1 = (const float*)d_in[25];
    float* out = (float*)d_out;

    // ---- workspace layout ----
    float* wsp = (float*)d_ws;
    size_t off = 0;
    auto walloc = [&](size_t n) { float* p = wsp + off; off += n; return p; };
    float* col   = walloc(65536ull * 576);   // largest im2col (conv2)
    float* actA  = walloc(262144ull * 64);   // act1 / act3 / fc1
    float* actB  = walloc(65536ull * 128);   // act2 / feat / fc2
    float* enc   = walloc(1024ull * ZDIM);
    float* hbuf  = walloc(2ull * BATCH * HDIM);
    float* cbuf  = walloc(2ull * BATCH * HDIM);
    float* mem   = walloc((size_t)BATCH * MSLOTS * WWID);
    float* link  = walloc((size_t)BATCH * MSLOTS * MSLOTS);
    float* prec  = walloc((size_t)BATCH * MSLOTS);
    float* rwb   = walloc((size_t)BATCH * RHEADS * MSLOTS);
    float* wwb   = walloc((size_t)BATCH * MSLOTS);
    float* usage = walloc((size_t)BATCH * MSLOTS);
    float* rvec  = walloc((size_t)BATCH * RHEADS * WWID);
    float* xbuf  = walloc((size_t)BATCH * NN_IN);
    float* gates = walloc((size_t)BATCH * 4 * HDIM);
    float* oclip = walloc((size_t)BATCH * HDIM);
    float* vbuf  = walloc((size_t)BATCH * IFACE);
    float* allc  = walloc((size_t)BATCH * MSLOTS);
    float* wcw   = walloc((size_t)BATCH * MSLOTS);
    float* rcw   = walloc((size_t)BATCH * RHEADS * MSLOTS);
    float* fw    = walloc((size_t)BATCH * RHEADS * MSLOTS);
    float* bw    = walloc((size_t)BATCH * RHEADS * MSLOTS);
    float* sumwv = walloc(BATCH);
    float* yin   = walloc((size_t)BATCH * 768);
    float* probe = walloc(32);

    // f16 weight region
    _Float16* hbase = (_Float16*)(wsp + off);
    size_t hoff = 0;
    auto halloc = [&](size_t n) { _Float16* p = hbase + hoff; hoff += n; return p; };
    _Float16* c1h  = halloc(64ull * 27);
    _Float16* c2h  = halloc(128ull * 576);
    _Float16* c3h  = halloc(256ull * 1152);
    _Float16* l1h  = halloc(2048ull * 4096);
    _Float16* l2h  = halloc(1024ull * 2048);
    _Float16* l3h  = halloc(256ull * 1024);
    _Float16* outh = halloc(256ull * 768);
    _Float16* ifh  = halloc((size_t)IFACE * HDIM);
    _Float16* Wih0h = halloc(2048ull * NN_IN);
    _Float16* Whh0h = halloc(2048ull * HDIM);
    _Float16* Wih1h = halloc(2048ull * HDIM);
    _Float16* Whh1h = halloc(2048ull * HDIM);

    // CDNA5 async-path probe (once)
    cdna5_async_probe_kernel<<<1, 32, 0, stream>>>(input, probe);

    // ---- convert all weights to f16 ----
    cvt16(stream, c1w, c1h, 64 * 27);
    cvt16(stream, c2w, c2h, 128 * 576);
    cvt16(stream, c3w, c3h, 256 * 1152);
    cvt16(stream, l1w, l1h, 2048 * 4096);
    cvt16(stream, l2w, l2h, 1024 * 2048);
    cvt16(stream, l3w, l3h, 256 * 1024);
    cvt16(stream, outw, outh, 256 * 768);
    cvt16(stream, ifw, ifh, IFACE * HDIM);
    cvt16(stream, Wih0, Wih0h, 2048 * NN_IN);
    cvt16(stream, Whh0, Whh0h, 2048 * HDIM);
    cvt16(stream, Wih1, Wih1h, 2048 * HDIM);
    cvt16(stream, Whh1, Whh1h, 2048 * HDIM);

    // ================= Phase A: conv + MLP encoder (1024 images) ===========
    im2col_nchw_kernel<<<gr(1024ll * 256 * 27), 256, 0, stream>>>(input, col, 1024, 3, 32, 32, 16, 16);
    gemm(stream, col, 27, c1h, 27, c1b, actA, 64, 262144, 64, 27, 0, 1, 0.01f);
    im2col_nhwc_kernel<<<gr(1024ll * 64 * 576), 256, 0, stream>>>(actA, col, 1024, 64, 16, 16, 8, 8);
    gemm(stream, col, 576, c2h, 576, c2b, actB, 128, 65536, 128, 576, 0, 1, 0.01f);
    im2col_nhwc_kernel<<<gr(1024ll * 16 * 1152), 256, 0, stream>>>(actB, col, 1024, 128, 8, 8, 4, 4);
    gemm(stream, col, 1152, c3h, 1152, c3b, actA, 256, 16384, 256, 1152, 0, 1, 0.01f);
    nhwc_to_flat_kernel<<<gr(1024ll * 4096), 256, 0, stream>>>(actA, actB);
    gemm(stream, actB, 4096, l1h, 4096, l1b, actA, 2048, 1024, 2048, 4096, 0, 0, 0.f);
    gemm(stream, actA, 2048, l2h, 2048, l2b, actB, 1024, 1024, 1024, 2048, 0, 0, 0.f);
    gemm(stream, actB, 1024, l3h, 1024, l3b, enc, ZDIM, 1024, ZDIM, 1024, 0, 2, 0.f);

    // ================= Phase B: recurrent LSTM + DNC memory ================
    hipMemcpyAsync(hbuf, h0, sizeof(float) * 2 * BATCH * HDIM, hipMemcpyDeviceToDevice, stream);
    hipMemcpyAsync(cbuf, h0, sizeof(float) * 2 * BATCH * HDIM, hipMemcpyDeviceToDevice, stream);
    fill_kernel<<<gr(BATCH * MSLOTS * WWID), 256, 0, stream>>>(mem, 1e-6f, BATCH * MSLOTS * WWID);
    fill_kernel<<<gr((long long)BATCH * MSLOTS * MSLOTS), 256, 0, stream>>>(link, 0.f, BATCH * MSLOTS * MSLOTS);
    fill_kernel<<<gr(BATCH * MSLOTS), 256, 0, stream>>>(prec, 0.f, BATCH * MSLOTS);
    fill_kernel<<<gr(BATCH * RHEADS * MSLOTS), 256, 0, stream>>>(rwb, 0.f, BATCH * RHEADS * MSLOTS);
    fill_kernel<<<gr(BATCH * MSLOTS), 256, 0, stream>>>(wwb, 0.f, BATCH * MSLOTS);
    fill_kernel<<<gr(BATCH * MSLOTS), 256, 0, stream>>>(usage, 0.f, BATCH * MSLOTS);
    fill_kernel<<<gr(BATCH * RHEADS * WWID), 256, 0, stream>>>(rvec, 0.f, BATCH * RHEADS * WWID);

    float* h0p = hbuf;                 float* h1p = hbuf + BATCH * HDIM;
    float* c0p = cbuf;                 float* c1p = cbuf + BATCH * HDIM;

    for (int t = 0; t < TSTEPS; ++t) {
        build_x_kernel<<<gr(BATCH * NN_IN), 256, 0, stream>>>(xbuf, enc, rvec, t);
        // layer 0: gates = x@Wih0^T + bih0 + h0@Whh0^T + bhh0
        gemm(stream, xbuf, NN_IN, Wih0h, NN_IN, bih0, gates, 4 * HDIM, BATCH, 4 * HDIM, NN_IN, 0, 0, 0.f);
        gemm(stream, h0p, HDIM, Whh0h, HDIM, bhh0, gates, 4 * HDIM, BATCH, 4 * HDIM, HDIM, 1, 0, 0.f);
        lstm_pointwise_kernel<<<gr(BATCH * HDIM), 256, 0, stream>>>(gates, h0p, c0p);
        // layer 1
        gemm(stream, h0p, HDIM, Wih1h, HDIM, bih1, gates, 4 * HDIM, BATCH, 4 * HDIM, HDIM, 0, 0, 0.f);
        gemm(stream, h1p, HDIM, Whh1h, HDIM, bhh1, gates, 4 * HDIM, BATCH, 4 * HDIM, HDIM, 1, 0, 0.f);
        lstm_pointwise_kernel<<<gr(BATCH * HDIM), 256, 0, stream>>>(gates, h1p, c1p);
        clip_kernel<<<gr(BATCH * HDIM), 256, 0, stream>>>(h1p, oclip);
        // interface vector
        gemm(stream, oclip, HDIM, ifh, HDIM, ifb, vbuf, IFACE, BATCH, IFACE, HDIM, 0, 0, 0.f);
        // memory step
        usage_kernel<<<gr(BATCH * MSLOTS), 256, 0, stream>>>(usage, wwb, rwb, vbuf);
        alloc_kernel<<<BATCH, MSLOTS, 0, stream>>>(usage, allc);
        content_kernel<<<BATCH, MSLOTS, 0, stream>>>(mem, vbuf, 260, 324, 1, wcw);   // write key
        ww_kernel<<<gr(BATCH * MSLOTS), 256, 0, stream>>>(wwb, vbuf, allc, wcw);
        link_kernel<<<gr((long long)BATCH * MSLOTS * MSLOTS), 256, 0, stream>>>(link, wwb, prec);
        sumwv_kernel<<<BATCH, MSLOTS, 0, stream>>>(wwb, sumwv);
        mem_update_kernel<<<gr(BATCH * MSLOTS * WWID), 256, 0, stream>>>(mem, wwb, vbuf);
        prec_kernel<<<gr(BATCH * MSLOTS), 256, 0, stream>>>(prec, wwb, sumwv);
        fwbw_kernel<<<gr(BATCH * RHEADS * MSLOTS), 256, 0, stream>>>(rwb, link, fw, bw);
        content_kernel<<<BATCH * RHEADS, MSLOTS, 0, stream>>>(mem, vbuf, 0, 256, RHEADS, rcw);
        rwnew_kernel<<<gr(BATCH * RHEADS * MSLOTS), 256, 0, stream>>>(rwb, vbuf, bw, rcw, fw);
        readvec_kernel<<<gr(BATCH * RHEADS * WWID), 256, 0, stream>>>(rwb, mem, rvec);
        // y[:, t] = concat(out*0.2, read_vecs) @ out_w^T + out_b
        build_y_kernel<<<gr(BATCH * 768), 256, 0, stream>>>(yin, oclip, rvec);
        gemm(stream, yin, 768, outh, 768, outb, out + (size_t)t * ZDIM, TSTEPS * ZDIM,
             BATCH, ZDIM, 768, 0, 0, 0.f);
    }
}
